// Block_59493886984599
// MI455X (gfx1250) — compile-verified, matrix operand order checked by reading, not run on hardware
//
#include <hip/hip_runtime.h>
#include <hip/hip_bf16.h>

// ---------------------------------------------------------------------------
// Problem constants (B=1)
// ---------------------------------------------------------------------------
static constexpr int S_   = 2048;
static constexpr int D_   = 1024;
static constexpr int H_   = 16;
static constexpr int HD_  = 64;     // head dim
static constexpr int HID_ = 4096;
static constexpr int E_   = 8;
static constexpr int TK_  = 2;
static constexpr int T_   = S_;     // tokens

typedef __bf16 bf16_t;
typedef __attribute__((ext_vector_type(16))) __bf16 v16bf;
typedef __attribute__((ext_vector_type(8)))  __bf16 v8bf;
typedef __attribute__((ext_vector_type(4)))  __bf16 v4bf;
typedef __attribute__((ext_vector_type(8)))  float  v8f;
typedef unsigned int u32;
typedef __attribute__((ext_vector_type(4))) u32 u32x4;
typedef __attribute__((ext_vector_type(8))) int i32x8;
typedef __attribute__((ext_vector_type(4))) int i32x4;

__device__ __forceinline__ v16bf frag16(const bf16_t* p0, const bf16_t* p1) {
  // Build a 16x(K-half) WMMA fragment from two contiguous 16B chunks.
  v8bf a = *(const v8bf*)p0;
  v8bf b = *(const v8bf*)p1;
  return __builtin_shufflevector(a, b, 0,1,2,3,4,5,6,7,8,9,10,11,12,13,14,15);
}
__device__ __forceinline__ v8f vzero8() {
  v8f z = {0.f,0.f,0.f,0.f,0.f,0.f,0.f,0.f};
  return z;
}
#define WMMA_BF16(acc_, a_, b_)                                              \
  (acc_) = __builtin_amdgcn_wmma_f32_16x16x32_bf16(false, (a_), false, (b_), \
                                                   (short)0, (acc_), false, false)

// ---------------------------------------------------------------------------
// Tensor Data Mover: DMA a [tile_d1 x 32-half] bf16 tile (row stride =
// stride_elems halfs) from global into LDS, padding each 64B row with 16B
// (LDS row pitch = 80B = LDT) so WMMA fragment reads stay bank-friendly.
// D# layout per CDNA5 ISA (§8.3/§8.4): count=1, type=2,
// data_size=2B, pad_interval=16 DWORDs (enc 3), pad_amount=4 DWORDs (enc 3).
// Issued by one wave per workgroup; completion via TENSORcnt.
// This toolchain's builtin takes 6 args: (g0, g1, g2, g3, g4, cpol).
// ---------------------------------------------------------------------------
__device__ __forceinline__ void tdm_load_tile32(u32 lds_addr, const void* gaddr,
                                                u32 tile_d1, u32 stride_elems) {
  const unsigned long long ga = (unsigned long long)(uintptr_t)gaddr;
  u32x4 g0;
  g0.x = 1u;                                            // count=1 (valid D#)
  g0.y = lds_addr;                                      // LDS byte address
  g0.z = (u32)ga;                                       // global_addr[31:0]
  g0.w = (u32)((ga >> 32) & 0x01FFFFFFu) | (2u << 30);  // addr[56:32] | type=2
  i32x8 g1;
  const u32 w0 = (1u << 16)   // data_size = 2 bytes
               | (1u << 20)   // pad_enable
               | (3u << 22)   // pad_interval: 16 DWORDs (64B) between pads
               | (3u << 25);  // pad_amount: 4 DWORDs (16B)
  const u32 td0 = stride_elems;     // tensor_dim0 (>= tile_dim0, no OOB)
  const u32 td1 = tile_d1;          // tensor_dim1
  g1[0] = (int)w0;
  g1[1] = (int)((td0 & 0xFFFFu) << 16);                   // tensor_dim0 lo16
  g1[2] = (int)((td0 >> 16) | ((td1 & 0xFFFFu) << 16));   // td0 hi / td1 lo
  g1[3] = (int)((td1 >> 16) | (32u << 16));               // td1 hi | tile_dim0=32
  g1[4] = (int)(tile_d1 & 0xFFFFu);                       // tile_dim1 (tile_dim2=0)
  g1[5] = (int)(u32)stride_elems;                         // tensor_dim0_stride lo32
  g1[6] = 0;                                              // stride hi16 | dim1_stride
  g1[7] = 0;
  const i32x4 z4 = {0, 0, 0, 0};
  const i32x8 z8 = {0, 0, 0, 0, 0, 0, 0, 0};
  __builtin_amdgcn_tensor_load_to_lds(g0, g1, z4, z4, z8, 0);
}
__device__ __forceinline__ u32 lds_addr_of(const void* p) {
  // Flat shared-aperture addresses carry the LDS byte offset in the low bits.
  return (u32)(uintptr_t)p;
}

// ---------------------------------------------------------------------------
// LayerNorm (f32 in -> bf16 out), one row of D=1024 per 256-thread block
// ---------------------------------------------------------------------------
__global__ __launch_bounds__(256) void ln_kernel(const float* __restrict__ x,
                                                 const float* __restrict__ g,
                                                 const float* __restrict__ b,
                                                 bf16_t* __restrict__ out) {
  const int row = blockIdx.x;
  const int tid = threadIdx.x;
  const float4 v = ((const float4*)(x + (size_t)row * D_))[tid];
  float s  = v.x + v.y + v.z + v.w;
  float s2 = v.x*v.x + v.y*v.y + v.z*v.z + v.w*v.w;
#pragma unroll
  for (int m = 16; m >= 1; m >>= 1) {
    s  += __shfl_xor(s,  m);
    s2 += __shfl_xor(s2, m);
  }
  __shared__ float ssum[8], ssum2[8], mu_s, rs_s;
  const int wid = tid >> 5, lane = tid & 31;
  if (lane == 0) { ssum[wid] = s; ssum2[wid] = s2; }
  __syncthreads();
  if (tid == 0) {
    float a = 0.f, a2 = 0.f;
#pragma unroll
    for (int i = 0; i < 8; i++) { a += ssum[i]; a2 += ssum2[i]; }
    const float mu  = a * (1.f / D_);
    const float var = a2 * (1.f / D_) - mu * mu;
    mu_s = mu;
    rs_s = rsqrtf(var + 1e-5f);
  }
  __syncthreads();
  const float mu = mu_s, rs = rs_s;
  const float4 gg = ((const float4*)g)[tid];
  const float4 bb = ((const float4*)b)[tid];
  v4bf o = { (bf16_t)((v.x - mu) * rs * gg.x + bb.x),
             (bf16_t)((v.y - mu) * rs * gg.y + bb.y),
             (bf16_t)((v.z - mu) * rs * gg.z + bb.z),
             (bf16_t)((v.w - mu) * rs * gg.w + bb.w) };
  ((v4bf*)(out + (size_t)row * D_))[tid] = o;
}

// ---------------------------------------------------------------------------
// f32 [K,N] -> bf16 [N,K] transpose+convert (32x32 LDS tile), batched over z
// ---------------------------------------------------------------------------
__global__ __launch_bounds__(256) void transpose_cvt_kernel(
    const float* __restrict__ in, bf16_t* __restrict__ out, int K, int N) {
  __shared__ float tile[32][33];
  const size_t bo = (size_t)blockIdx.z * (size_t)K * (size_t)N;
  in  += bo;
  out += bo;
  const int tx = threadIdx.x, ty = threadIdx.y;       // (32,8)
  const int n0 = blockIdx.x * 32, k0 = blockIdx.y * 32;
#pragma unroll
  for (int i = 0; i < 4; i++) {
    const int k = k0 + ty + i * 8;
    tile[ty + i * 8][tx] = in[(size_t)k * N + n0 + tx];
  }
  __syncthreads();
#pragma unroll
  for (int i = 0; i < 4; i++) {
    const int n = n0 + ty + i * 8;
    out[(size_t)n * K + k0 + tx] = (bf16_t)tile[tx][ty + i * 8];
  }
}

// ---------------------------------------------------------------------------
// Generic WMMA GEMM: C[M,N] = A[M,K] * Bt[N,K]^T   (A, Bt bf16 row-major)
// Block tile 128x128, 8 waves, wave tile 64x32 (4 Mx 2 N 16x16 tiles).
// A/B tiles DMA'd by the Tensor Data Mover into double-buffered LDS; the
// DMA for k-step t+1 overlaps WMMA on k-step t (wait TENSORcnt<=2).
// STORE_BF16: write bf16 to Cb; else f32 to Cf (optionally + resid).
// M, N multiples of 128; K multiple of 32.
// ---------------------------------------------------------------------------
template <bool STORE_BF16, bool RESID>
__global__ __launch_bounds__(256) void gemm_bt_kernel(
    const bf16_t* __restrict__ A, const bf16_t* __restrict__ Bt,
    float* __restrict__ Cf, bf16_t* __restrict__ Cb,
    const float* __restrict__ resid, int M, int N, int K) {
  constexpr int LDT = 40;                       // halfs (80B row, 16B aligned)
  __shared__ bf16_t Asm[2][128 * LDT];
  __shared__ bf16_t Bsm[2][128 * LDT];
  const int tid  = threadIdx.x;
  const int wid  = tid >> 5, lane = tid & 31;
  const int wm   = wid & 1, wn = wid >> 1;      // wave: rows wm*64, cols wn*32
  const int mrow = lane & 15, khalf = lane >> 4;
  const int row0 = blockIdx.y * 128, col0 = blockIdx.x * 128;

  v8f acc[4][2];
#pragma unroll
  for (int mt = 0; mt < 4; mt++)
#pragma unroll
    for (int nt = 0; nt < 2; nt++) acc[mt][nt] = vzero8();

  const bf16_t* Ab = A + (size_t)row0 * K;
  const bf16_t* Bb = Bt + (size_t)col0 * K;
  const int nk = K >> 5;
  if (wid == 0) {
    tdm_load_tile32(lds_addr_of(Asm[0]), Ab, 128, (u32)K);
    tdm_load_tile32(lds_addr_of(Bsm[0]), Bb, 128, (u32)K);
  }
  for (int kt = 0; kt < nk; kt++) {
    if (wid == 0) {
      if (kt + 1 < nk) {
        tdm_load_tile32(lds_addr_of(Asm[(kt + 1) & 1]), Ab + (kt + 1) * 32, 128,
                        (u32)K);
        tdm_load_tile32(lds_addr_of(Bsm[(kt + 1) & 1]), Bb + (kt + 1) * 32, 128,
                        (u32)K);
        __builtin_amdgcn_s_wait_tensorcnt(2);   // tile kt complete (in-order)
      } else {
        __builtin_amdgcn_s_wait_tensorcnt(0);
      }
    }
    __syncthreads();
    const bf16_t* As = Asm[kt & 1];
    const bf16_t* Bs = Bsm[kt & 1];
    v16bf af[4], bfr[2];
#pragma unroll
    for (int mt = 0; mt < 4; mt++) {
      const bf16_t* p = As + (wm * 64 + mt * 16 + mrow) * LDT + khalf * 8;
      af[mt] = frag16(p, p + 16);
    }
#pragma unroll
    for (int nt = 0; nt < 2; nt++) {
      const bf16_t* p = Bs + (wn * 32 + nt * 16 + mrow) * LDT + khalf * 8;
      bfr[nt] = frag16(p, p + 16);
    }
#pragma unroll
    for (int mt = 0; mt < 4; mt++)
#pragma unroll
      for (int nt = 0; nt < 2; nt++) WMMA_BF16(acc[mt][nt], af[mt], bfr[nt]);
    __syncthreads();
  }

  const int hl = lane >> 4, nl = lane & 15;
#pragma unroll
  for (int mt = 0; mt < 4; mt++)
#pragma unroll
    for (int nt = 0; nt < 2; nt++)
#pragma unroll
      for (int r = 0; r < 8; r++) {
        const int row = row0 + wm * 64 + mt * 16 + r + 8 * hl;
        const int col = col0 + wn * 32 + nt * 16 + nl;
        float c = acc[mt][nt][r];
        if (RESID) c += resid[(size_t)row * N + col];
        if (STORE_BF16) Cb[(size_t)row * N + col] = (bf16_t)c;
        else            Cf[(size_t)row * N + col] = c;
      }
}

// ---------------------------------------------------------------------------
// Flash attention (causal). Grid: (S/128, H). 8 waves x 16 query rows.
// Q,K fragments straight from global (layout-contiguous); V transposed in LDS.
// ---------------------------------------------------------------------------
__global__ __launch_bounds__(256) void flash_attn_kernel(
    const bf16_t* __restrict__ Q, const bf16_t* __restrict__ Km,
    const bf16_t* __restrict__ V, bf16_t* __restrict__ O) {
  constexpr int VLD = 72;                       // halfs (144B rows, 16B aligned)
  __shared__ bf16_t Vt[HD_ * VLD];              // V^T tile: [d][key]
  __shared__ bf16_t Pl[8 * 16 * VLD];           // per-wave P tile: [row][key]
  const int tid = threadIdx.x, wid = tid >> 5, lane = tid & 31;
  const int h = blockIdx.y, qb = blockIdx.x;
  const int qbase = qb * 128 + wid * 16;
  const int mrow = lane & 15, khalf = lane >> 4;
  const int hl = khalf, nl = mrow;

  v16bf qf[2];
  {
    const bf16_t* qp = Q + (size_t)(qbase + mrow) * D_ + h * HD_;
#pragma unroll
    for (int ds = 0; ds < 2; ds++) {
      const bf16_t* p = qp + ds * 32 + khalf * 8;
      qf[ds] = frag16(p, p + 16);
    }
  }

  float mrun[8], lrun[8];
  v8f oacc[4];
#pragma unroll
  for (int r = 0; r < 8; r++) { mrun[r] = -1e30f; lrun[r] = 0.f; }
#pragma unroll
  for (int nt = 0; nt < 4; nt++) oacc[nt] = vzero8();

  const int kb_end = qb * 2 + 2;                // key tiles of 64 up to diagonal
  for (int kb = 0; kb < kb_end; kb++) {
    {   // cooperative V^T staging: 64 keys x 64 d
      const int key = tid >> 2;
      const int d0  = (tid & 3) * 16;
      const bf16_t* vp = V + (size_t)(kb * 64 + key) * D_ + h * HD_ + d0;
      v8bf a = *(const v8bf*)vp;
      v8bf b = *(const v8bf*)(vp + 8);
#pragma unroll
      for (int i = 0; i < 8; i++) Vt[(d0 + i) * VLD + key] = a[i];
#pragma unroll
      for (int i = 0; i < 8; i++) Vt[(d0 + 8 + i) * VLD + key] = b[i];
    }
    __syncthreads();

    // S = (Q K^T) / sqrt(HD), causal-masked
    v8f st[4];
#pragma unroll
    for (int nt = 0; nt < 4; nt++) {
      st[nt] = vzero8();
#pragma unroll
      for (int ds = 0; ds < 2; ds++) {
        const bf16_t* kp = Km + (size_t)(kb * 64 + nt * 16 + nl) * D_ + h * HD_ +
                           ds * 32 + khalf * 8;
        v16bf kf = frag16(kp, kp + 16);
        WMMA_BF16(st[nt], qf[ds], kf);
      }
      const int kj = kb * 64 + nt * 16 + nl;
#pragma unroll
      for (int r = 0; r < 8; r++) {
        const int qi = qbase + r + 8 * hl;
        const float sv = st[nt][r] * 0.125f;
        st[nt][r] = (kj > qi) ? -1e30f : sv;
      }
    }

    // online softmax (16-lane row reductions within each wave half)
    float mnew[8], alpha[8], rsum[8];
#pragma unroll
    for (int r = 0; r < 8; r++) {
      float v = fmaxf(fmaxf(st[0][r], st[1][r]), fmaxf(st[2][r], st[3][r]));
#pragma unroll
      for (int m = 1; m < 16; m <<= 1) v = fmaxf(v, __shfl_xor(v, m));
      mnew[r]  = fmaxf(mrun[r], v);
      alpha[r] = exp2f((mrun[r] - mnew[r]) * 1.44269504f);
      mrun[r]  = mnew[r];
      rsum[r]  = 0.f;
    }
    bf16_t* pw = Pl + wid * 16 * VLD;
#pragma unroll
    for (int nt = 0; nt < 4; nt++)
#pragma unroll
      for (int r = 0; r < 8; r++) {
        const float sv = st[nt][r];
        const float p  = (sv <= -1e29f) ? 0.f
                                        : exp2f((sv - mnew[r]) * 1.44269504f);
        rsum[r] += p;
        pw[(r + 8 * hl) * VLD + nt * 16 + nl] = (bf16_t)p;
      }
#pragma unroll
    for (int r = 0; r < 8; r++) {
      float v = rsum[r];
#pragma unroll
      for (int m = 1; m < 16; m <<= 1) v += __shfl_xor(v, m);
      lrun[r] = lrun[r] * alpha[r] + v;
    }
#pragma unroll
    for (int nt = 0; nt < 4; nt++)
#pragma unroll
      for (int r = 0; r < 8; r++) oacc[nt][r] *= alpha[r];

    // O += P V
#pragma unroll
    for (int kk = 0; kk < 2; kk++) {
      const bf16_t* pp = pw + mrow * VLD + kk * 32 + khalf * 8;
      v16bf pf = frag16(pp, pp + 16);
#pragma unroll
      for (int nt = 0; nt < 4; nt++) {
        const bf16_t* vp2 = Vt + (nt * 16 + nl) * VLD + kk * 32 + khalf * 8;
        v16bf vf = frag16(vp2, vp2 + 16);
        WMMA_BF16(oacc[nt], pf, vf);
      }
    }
    __syncthreads();
  }

#pragma unroll
  for (int nt = 0; nt < 4; nt++)
#pragma unroll
    for (int r = 0; r < 8; r++) {
      const int q = qbase + r + 8 * hl;
      const int d = nt * 16 + nl;
      const float inv = (lrun[r] > 0.f) ? (1.f / lrun[r]) : 0.f;
      O[(size_t)q * D_ + h * HD_ + d] = (bf16_t)(oacc[nt][r] * inv);
    }
}

// ---------------------------------------------------------------------------
// Router: sigmoid(h2 @ w_gate), top-2 normalized weights. One wave per token.
// ---------------------------------------------------------------------------
__global__ __launch_bounds__(256) void router_kernel(
    const bf16_t* __restrict__ h2, const float* __restrict__ wg,
    int* __restrict__ tki, float* __restrict__ tkw) {
  const int wid = threadIdx.x >> 5, lane = threadIdx.x & 31;
  const int tok = blockIdx.x * 8 + wid;
  float acc[E_];
#pragma unroll
  for (int e = 0; e < E_; e++) acc[e] = 0.f;
  const bf16_t* xr = h2 + (size_t)tok * D_;
  for (int d = lane; d < D_; d += 32) {
    const float x = (float)xr[d];
    const float* wr = wg + d * E_;
#pragma unroll
    for (int e = 0; e < E_; e++) acc[e] += x * wr[e];
  }
#pragma unroll
  for (int e = 0; e < E_; e++)
#pragma unroll
    for (int m = 16; m >= 1; m >>= 1) acc[e] += __shfl_xor(acc[e], m);
  if (lane == 0) {
    float s[E_];
#pragma unroll
    for (int e = 0; e < E_; e++) s[e] = 1.f / (1.f + __expf(-acc[e]));
    int i1 = 0;
#pragma unroll
    for (int e = 1; e < E_; e++) if (s[e] > s[i1]) i1 = e;
    int i2 = (i1 == 0) ? 1 : 0;
#pragma unroll
    for (int e = 0; e < E_; e++) if (e != i1 && s[e] > s[i2]) i2 = e;
    const float ws = s[i1] + s[i2] + 1e-9f;
    tki[tok * 2 + 0] = i1;  tkw[tok * 2 + 0] = s[i1] / ws;
    tki[tok * 2 + 1] = i2;  tkw[tok * 2 + 1] = s[i2] / ws;
  }
}

// ---------------------------------------------------------------------------
// Compact routing into per-expert token/weight lists (single block).
// ---------------------------------------------------------------------------
__global__ __launch_bounds__(256) void compact_kernel(
    const int* __restrict__ tki, const float* __restrict__ tkw,
    int* __restrict__ etok, float* __restrict__ ew,
    int* __restrict__ eoff, int* __restrict__ ecnt) {
  __shared__ int cnt[E_], off[E_], pos[E_];
  const int tid = threadIdx.x;
  if (tid < E_) cnt[tid] = 0;
  __syncthreads();
  for (int a = tid; a < T_ * TK_; a += 256) atomicAdd(&cnt[tki[a]], 1);
  __syncthreads();
  if (tid == 0) {
    int o = 0;
    for (int e = 0; e < E_; e++) { off[e] = o; o += cnt[e]; }
  }
  __syncthreads();
  if (tid < E_) { pos[tid] = 0; eoff[tid] = off[tid]; ecnt[tid] = cnt[tid]; }
  __syncthreads();
  for (int a = tid; a < T_ * TK_; a += 256) {
    const int e = tki[a];
    const int p = atomicAdd(&pos[e], 1);
    etok[off[e] + p] = a >> 1;
    ew[off[e] + p]   = tkw[a];
  }
}

// ---------------------------------------------------------------------------
// Gathered expert gate+up GEMM, fused SiLU(g)*u -> act (bf16).
// Grid: (HID/64, T/128, E). Block tile 128x64, wave tile 64x16 (dual acc).
// Wg/Wu tiles via TDM double-buffer; gathered A rows via vector loads.
// ---------------------------------------------------------------------------
__global__ __launch_bounds__(256) void moe_gateup_kernel(
    const bf16_t* __restrict__ h2, const bf16_t* __restrict__ WgT,
    const bf16_t* __restrict__ WuT, const int* __restrict__ etok,
    const int* __restrict__ eoff, const int* __restrict__ ecnt,
    bf16_t* __restrict__ act) {
  constexpr int LDT = 40;
  __shared__ bf16_t Asm[128 * LDT];
  __shared__ bf16_t Gsm[2][64 * LDT];
  __shared__ bf16_t Usm[2][64 * LDT];
  __shared__ int idx[128];
  const int e = blockIdx.z;
  const int cnt = ecnt[e];
  const int row0 = blockIdx.y * 128;
  if (row0 >= cnt) return;
  const int off = eoff[e];
  const int col0 = blockIdx.x * 64;
  const int tid = threadIdx.x, wid = tid >> 5, lane = tid & 31;
  if (tid < 128) {
    const int r = row0 + tid;
    idx[tid] = etok[off + (r < cnt ? r : 0)];
  }
  __syncthreads();
  const bf16_t* Wg = WgT + (size_t)e * HID_ * D_ + (size_t)col0 * D_;
  const bf16_t* Wu = WuT + (size_t)e * HID_ * D_ + (size_t)col0 * D_;
  const int wm = wid & 1, wn = wid >> 1;        // rows wm*64, cols wn*16
  const int mrow = lane & 15, khalf = lane >> 4;
  v8f ag[4], au[4];
#pragma unroll
  for (int mt = 0; mt < 4; mt++) { ag[mt] = vzero8(); au[mt] = vzero8(); }

  const int arow = tid >> 1;
  const int acho = (tid & 1) * 2;
  const int nk = D_ >> 5;
  if (wid == 0) {
    tdm_load_tile32(lds_addr_of(Gsm[0]), Wg, 64, (u32)D_);
    tdm_load_tile32(lds_addr_of(Usm[0]), Wu, 64, (u32)D_);
  }
  for (int kt = 0; kt < nk; kt++) {
    const int k0 = kt * 32;
    {   // gathered A tile (vector path)
      const uint4* pa =
          (const uint4*)(h2 + (size_t)idx[arow] * D_ + k0) + acho;
      uint4 x0 = pa[0], x1 = pa[1];
      uint4* da = (uint4*)(Asm + arow * LDT) + acho;
      da[0] = x0; da[1] = x1;
    }
    if (wid == 0) {
      if (kt + 1 < nk) {
        tdm_load_tile32(lds_addr_of(Gsm[(kt + 1) & 1]), Wg + (kt + 1) * 32, 64,
                        (u32)D_);
        tdm_load_tile32(lds_addr_of(Usm[(kt + 1) & 1]), Wu + (kt + 1) * 32, 64,
                        (u32)D_);
        __builtin_amdgcn_s_wait_tensorcnt(2);
      } else {
        __builtin_amdgcn_s_wait_tensorcnt(0);
      }
    }
    __syncthreads();
    v16bf bg, bu;
    {
      const bf16_t* p = Gsm[kt & 1] + (wn * 16 + mrow) * LDT + khalf * 8;
      bg = frag16(p, p + 16);
      const bf16_t* q = Usm[kt & 1] + (wn * 16 + mrow) * LDT + khalf * 8;
      bu = frag16(q, q + 16);
    }
#pragma unroll
    for (int mt = 0; mt < 4; mt++) {
      const bf16_t* p = Asm + (wm * 64 + mt * 16 + mrow) * LDT + khalf * 8;
      v16bf af = frag16(p, p + 16);
      WMMA_BF16(ag[mt], af, bg);
      WMMA_BF16(au[mt], af, bu);
    }
    __syncthreads();
  }

  const int hl = lane >> 4, nl = lane & 15;
#pragma unroll
  for (int mt = 0; mt < 4; mt++)
#pragma unroll
    for (int r = 0; r < 8; r++) {
      const int lr = wm * 64 + mt * 16 + r + 8 * hl;
      if (row0 + lr < cnt) {
        const float g = ag[mt][r], u = au[mt][r];
        const float a = g / (1.f + __expf(-g)) * u;   // silu(g)*u
        act[(size_t)(off + row0 + lr) * HID_ + col0 + wn * 16 + nl] = (bf16_t)a;
      }
    }
}

// ---------------------------------------------------------------------------
// Expert down GEMM + weighted atomic scatter into out (residual pre-loaded).
// Grid: (D/128, T/128, E). Block tile 128x128, wave tile 64x32. K = HID.
// A (contiguous expert rows) and Wd tiles via TDM double-buffer.
// ---------------------------------------------------------------------------
__global__ __launch_bounds__(256) void moe_down_kernel(
    const bf16_t* __restrict__ act, const bf16_t* __restrict__ WdT,
    const int* __restrict__ etok, const float* __restrict__ ew,
    const int* __restrict__ eoff, const int* __restrict__ ecnt,
    float* __restrict__ out) {
  constexpr int LDT = 40;
  __shared__ bf16_t Asm[2][128 * LDT];
  __shared__ bf16_t Bsm[2][128 * LDT];
  __shared__ int stok[128];
  __shared__ float sw[128];
  const int e = blockIdx.z;
  const int cnt = ecnt[e];
  const int row0 = blockIdx.y * 128;
  if (row0 >= cnt) return;
  const int off = eoff[e];
  const int col0 = blockIdx.x * 128;
  const int tid = threadIdx.x, wid = tid >> 5, lane = tid & 31;
  if (tid < 128) {
    const int r = row0 + tid;
    const bool ok = r < cnt;
    stok[tid] = ok ? etok[off + r] : 0;
    sw[tid]   = ok ? ew[off + r] : 0.f;
  }
  __syncthreads();
  const bf16_t* Ab = act + (size_t)(off + row0) * HID_;
  const bf16_t* Bb = WdT + (size_t)e * D_ * HID_ + (size_t)col0 * HID_;
  const int wm = wid & 1, wn = wid >> 1;
  const int mrow = lane & 15, khalf = lane >> 4;
  v8f acc[4][2];
#pragma unroll
  for (int mt = 0; mt < 4; mt++)
#pragma unroll
    for (int nt = 0; nt < 2; nt++) acc[mt][nt] = vzero8();

  const int nk = HID_ >> 5;
  if (wid == 0) {
    tdm_load_tile32(lds_addr_of(Asm[0]), Ab, 128, (u32)HID_);
    tdm_load_tile32(lds_addr_of(Bsm[0]), Bb, 128, (u32)HID_);
  }
  for (int kt = 0; kt < nk; kt++) {
    if (wid == 0) {
      if (kt + 1 < nk) {
        tdm_load_tile32(lds_addr_of(Asm[(kt + 1) & 1]), Ab + (kt + 1) * 32, 128,
                        (u32)HID_);
        tdm_load_tile32(lds_addr_of(Bsm[(kt + 1) & 1]), Bb + (kt + 1) * 32, 128,
                        (u32)HID_);
        __builtin_amdgcn_s_wait_tensorcnt(2);
      } else {
        __builtin_amdgcn_s_wait_tensorcnt(0);
      }
    }
    __syncthreads();
    const bf16_t* As = Asm[kt & 1];
    const bf16_t* Bs = Bsm[kt & 1];
    v16bf af[4], bfr[2];
#pragma unroll
    for (int mt = 0; mt < 4; mt++) {
      const bf16_t* p = As + (wm * 64 + mt * 16 + mrow) * LDT + khalf * 8;
      af[mt] = frag16(p, p + 16);
    }
#pragma unroll
    for (int nt = 0; nt < 2; nt++) {
      const bf16_t* p = Bs + (wn * 32 + nt * 16 + mrow) * LDT + khalf * 8;
      bfr[nt] = frag16(p, p + 16);
    }
#pragma unroll
    for (int mt = 0; mt < 4; mt++)
#pragma unroll
      for (int nt = 0; nt < 2; nt++) WMMA_BF16(acc[mt][nt], af[mt], bfr[nt]);
    __syncthreads();
  }

  const int hl = lane >> 4, nl = lane & 15;
#pragma unroll
  for (int mt = 0; mt < 4; mt++)
#pragma unroll
    for (int nt = 0; nt < 2; nt++)
#pragma unroll
      for (int r = 0; r < 8; r++) {
        const int lr = wm * 64 + mt * 16 + r + 8 * hl;
        if (row0 + lr < cnt) {
          const int tok = stok[lr];
          const float w = sw[lr];
          atomicAdd(&out[(size_t)tok * D_ + col0 + wn * 32 + nt * 16 + nl],
                    acc[mt][nt][r] * w);
        }
      }
}

// ---------------------------------------------------------------------------
// Host orchestration
// ---------------------------------------------------------------------------
extern "C" void kernel_launch(void* const* d_in, const int* in_sizes, int n_in,
                              void* d_out, int out_size, void* d_ws,
                              size_t ws_size, hipStream_t stream) {
  (void)in_sizes; (void)n_in; (void)out_size; (void)ws_size;
  const float* x     = (const float*)d_in[0];
  const float* ln1g  = (const float*)d_in[1];
  const float* ln1b  = (const float*)d_in[2];
  const float* wq    = (const float*)d_in[3];
  const float* wk    = (const float*)d_in[4];
  const float* wv    = (const float*)d_in[5];
  const float* wo    = (const float*)d_in[6];
  const float* ln2g  = (const float*)d_in[7];
  const float* ln2b  = (const float*)d_in[8];
  const float* wgate = (const float*)d_in[9];
  const float* weg   = (const float*)d_in[10];
  const float* weu   = (const float*)d_in[11];
  const float* wed   = (const float*)d_in[12];
  float* out = (float*)d_out;                   // doubles as x1 = x + attn

  // Workspace bump allocator (256B aligned)
  char* p = (char*)d_ws;
  auto alloc = [&](size_t bytes) -> void* {
    void* r = (void*)p;
    p += (bytes + 255) & ~(size_t)255;
    return r;
  };
  bf16_t* wqT  = (bf16_t*)alloc((size_t)D_ * D_ * 2);
  bf16_t* wkT  = (bf16_t*)alloc((size_t)D_ * D_ * 2);
  bf16_t* wvT  = (bf16_t*)alloc((size_t)D_ * D_ * 2);
  bf16_t* woT  = (bf16_t*)alloc((size_t)D_ * D_ * 2);
  bf16_t* wegT = (bf16_t*)alloc((size_t)E_ * HID_ * D_ * 2);
  bf16_t* weuT = (bf16_t*)alloc((size_t)E_ * HID_ * D_ * 2);
  bf16_t* wedT = (bf16_t*)alloc((size_t)E_ * D_ * HID_ * 2);
  bf16_t* h1    = (bf16_t*)alloc((size_t)S_ * D_ * 2);
  bf16_t* qbuf  = (bf16_t*)alloc((size_t)S_ * D_ * 2);
  bf16_t* kbuf  = (bf16_t*)alloc((size_t)S_ * D_ * 2);
  bf16_t* vbuf  = (bf16_t*)alloc((size_t)S_ * D_ * 2);
  bf16_t* attnb = (bf16_t*)alloc((size_t)S_ * D_ * 2);
  bf16_t* h2    = (bf16_t*)alloc((size_t)S_ * D_ * 2);
  int*    tki  = (int*)alloc((size_t)T_ * TK_ * 4);
  float*  tkw  = (float*)alloc((size_t)T_ * TK_ * 4);
  int*    etok = (int*)alloc((size_t)T_ * TK_ * 4);
  float*  ewt  = (float*)alloc((size_t)T_ * TK_ * 4);
  int*    eoff = (int*)alloc(E_ * 4);
  int*    ecnt = (int*)alloc(E_ * 4);
  bf16_t* act  = (bf16_t*)alloc(((size_t)T_ * TK_ + 128) * HID_ * 2);  // +pad rows

  const dim3 tb(32, 8);
  // Weight convert + transpose to bf16 [N][K]
  transpose_cvt_kernel<<<dim3(D_/32, D_/32, 1), tb, 0, stream>>>(wq, wqT, D_, D_);
  transpose_cvt_kernel<<<dim3(D_/32, D_/32, 1), tb, 0, stream>>>(wk, wkT, D_, D_);
  transpose_cvt_kernel<<<dim3(D_/32, D_/32, 1), tb, 0, stream>>>(wv, wvT, D_, D_);
  transpose_cvt_kernel<<<dim3(D_/32, D_/32, 1), tb, 0, stream>>>(wo, woT, D_, D_);
  transpose_cvt_kernel<<<dim3(HID_/32, D_/32, E_), tb, 0, stream>>>(weg, wegT, D_, HID_);
  transpose_cvt_kernel<<<dim3(HID_/32, D_/32, E_), tb, 0, stream>>>(weu, weuT, D_, HID_);
  transpose_cvt_kernel<<<dim3(D_/32, HID_/32, E_), tb, 0, stream>>>(wed, wedT, HID_, D_);

  // LN1 -> h1 (bf16)
  ln_kernel<<<S_, 256, 0, stream>>>(x, ln1g, ln1b, h1);

  // QKV projections (bf16 out)
  gemm_bt_kernel<true, false><<<dim3(D_/128, S_/128), 256, 0, stream>>>(
      h1, wqT, nullptr, qbuf, nullptr, S_, D_, D_);
  gemm_bt_kernel<true, false><<<dim3(D_/128, S_/128), 256, 0, stream>>>(
      h1, wkT, nullptr, kbuf, nullptr, S_, D_, D_);
  gemm_bt_kernel<true, false><<<dim3(D_/128, S_/128), 256, 0, stream>>>(
      h1, wvT, nullptr, vbuf, nullptr, S_, D_, D_);

  // Causal flash attention
  flash_attn_kernel<<<dim3(S_/128, H_), 256, 0, stream>>>(qbuf, kbuf, vbuf, attnb);

  // O projection + residual: out = x + attn @ wo   (out == x1)
  gemm_bt_kernel<false, true><<<dim3(D_/128, S_/128), 256, 0, stream>>>(
      attnb, woT, out, nullptr, x, S_, D_, D_);

  // LN2 -> h2 (bf16)
  ln_kernel<<<S_, 256, 0, stream>>>(out, ln2g, ln2b, h2);

  // Routing: sigmoid top-2, normalized
  router_kernel<<<T_/8, 256, 0, stream>>>(h2, wgate, tki, tkw);
  compact_kernel<<<1, 256, 0, stream>>>(tki, tkw, etok, ewt, eoff, ecnt);

  // Expert gate+up (gathered) -> act = silu(g)*u
  moe_gateup_kernel<<<dim3(HID_/64, T_/128, E_), 256, 0, stream>>>(
      h2, wegT, weuT, etok, eoff, ecnt, act);

  // Expert down + weighted scatter-add into out (x1 already there)
  moe_down_kernel<<<dim3(D_/128, T_/128, E_), 256, 0, stream>>>(
      act, wedT, etok, ewt, eoff, ecnt, out);
}
